// NCDEtoODE_86157043958236
// MI455X (gfx1250) — matrix operand order repaired and checked by reading, not compiled, not run on hardware
//
#include <hip/hip_runtime.h>

typedef __attribute__((ext_vector_type(16))) _Float16 v16h;
typedef __attribute__((ext_vector_type(8)))  float    v8f;

namespace {
constexpr int BATCH = 512, LSEQ = 64, FEAT = 16, ZDIM = 32, CDIM = 8, HIDN = 64, HORZ = 12;
constexpr int BT = 16;            // batch rows per workgroup
constexpr int NW = 4;             // waves per workgroup
constexpr int NTHR = NW * 32;
constexpr int INW = ZDIM + CDIM;  // 40

// ---- LDS layout (bytes) ----
constexpr int OFF_W2C = 0;                                 // f16 swz cde_W2  [32 tiles][2 kk][32 lanes][16]
constexpr int OFF_XT  = OFF_W2C + 32 * 2 * 32 * 16 * 2;    // f16 x tile [16][64][16]
constexpr int OFF_W1C = OFF_XT  + BT * LSEQ * FEAT * 2;    // f16 swz cde_W1 [4][32][16]
constexpr int OFF_W1O = OFF_W1C + 4 * 32 * 16 * 2;
constexpr int OFF_W2O = OFF_W1O + 4 * 32 * 16 * 2;         // f16 swz ode_W2 [4][2][32][16]
constexpr int OFF_W3O = OFF_W2O + 4 * 2 * 32 * 16 * 2;     // f16 swz ode_W3 [2][2][32][16]
constexpr int OFF_EBC = OFF_W3O + 2 * 2 * 32 * 16 * 2;     // f32 eff-bias cde, C-layout [4][32][8]
constexpr int OFF_EBO = OFF_EBC + 4 * 32 * 8 * 4;
constexpr int OFF_B2C = OFF_EBO + 4 * 32 * 8 * 4;          // f32[512]
constexpr int OFF_B2O = OFF_B2C + 512 * 4;                 // f32[64]
constexpr int OFF_B3O = OFF_B2O + 64 * 4;                  // f32[32]
constexpr int OFF_WR  = OFF_B3O + 32 * 4;                  // f32[16][32]
constexpr int OFF_BR  = OFF_WR  + 512 * 4;                 // f32[16]
constexpr int OFF_Z   = OFF_BR  + 16 * 4;                  // f32[16][32]
constexpr int OFF_YST = OFF_Z   + 512 * 4;                 // f32[16][32]
constexpr int OFF_K   = OFF_YST + 512 * 4;                 // f32[6][16][32]
constexpr int OFF_HB  = OFF_K   + 6 * 512 * 4;             // f16[16][64]
constexpr int OFF_HB2 = OFF_HB  + BT * HIDN * 2;           // f16[16][64]
constexpr int SMEM_BYTES = OFF_HB2 + BT * HIDN * 2;        // ~148.4 KB

// ---- Tsit5 coefficients ----
constexpr float C2c = 0.161f, C3c = 0.327f, C4c = 0.9f, C5c = 0.9800255409045097f, C6c = 1.0f;
constexpr float A21 = 0.161f;
constexpr float A31 = -0.008480655492356989f, A32 = 0.335480655492357f;
constexpr float A41 = 2.8971530571054935f, A42 = -6.359448489975075f, A43 = 4.3622954328695815f;
constexpr float A51 = 5.325864828439257f, A52 = -11.748883564062828f, A53 = 7.4955393428898365f, A54 = -0.09249506636175525f;
constexpr float A61 = 5.86145544294642f, A62 = -12.92096931784711f, A63 = 8.159367898576159f, A64 = -0.071584973281401f, A65 = -0.028269050394068383f;
constexpr float Bc1 = 0.09646076681806523f, Bc2 = 0.01f, Bc3 = 0.4798896504144996f;
constexpr float Bc4 = 1.379008574103742f, Bc5 = -3.290069515436081f, Bc6 = 2.324710524099774f;
} // namespace

static __device__ __forceinline__ v8f wmma16(v16h a, v16h b, v8f c) {
  // D = A(16x32 f16) * B(32x16 f16) + C(16x16 f32)
  return __builtin_amdgcn_wmma_f32_16x16x32_f16(false, a, false, b, (short)0, c, false, false);
}

// fast tanh: v_exp_f32 + v_add + v_rcp_f32 + v_fma (no IEEE division chain)
static __device__ __forceinline__ float ftanh(float x) {
  float e = __expf(2.0f * x);
  float r = __builtin_amdgcn_rcpf(e + 1.0f);
  return __builtin_fmaf(-2.0f, r, 1.0f);
}

__global__ void __launch_bounds__(NTHR, 1)
ncde_persistent_kernel(const float* __restrict__ x_seq, const int* __restrict__ country_idx,
                       const float* __restrict__ cde_W1, const float* __restrict__ cde_b1,
                       const float* __restrict__ cde_W2, const float* __restrict__ cde_b2,
                       const float* __restrict__ ode_W1, const float* __restrict__ ode_b1,
                       const float* __restrict__ ode_W2, const float* __restrict__ ode_b2,
                       const float* __restrict__ ode_W3, const float* __restrict__ ode_b3,
                       const float* __restrict__ Wr, const float* __restrict__ br,
                       float* __restrict__ out) {
  extern __shared__ unsigned char smem_raw[];
  _Float16* sW2C = (_Float16*)(smem_raw + OFF_W2C);
  _Float16* sXT  = (_Float16*)(smem_raw + OFF_XT);
  _Float16* sW1C = (_Float16*)(smem_raw + OFF_W1C);
  _Float16* sW1O = (_Float16*)(smem_raw + OFF_W1O);
  _Float16* sW2O = (_Float16*)(smem_raw + OFF_W2O);
  _Float16* sW3O = (_Float16*)(smem_raw + OFF_W3O);
  float* sEBC = (float*)(smem_raw + OFF_EBC);
  float* sEBO = (float*)(smem_raw + OFF_EBO);
  float* sB2C = (float*)(smem_raw + OFF_B2C);
  float* sB2O = (float*)(smem_raw + OFF_B2O);
  float* sB3O = (float*)(smem_raw + OFF_B3O);
  float* sWR  = (float*)(smem_raw + OFF_WR);
  float* sBR  = (float*)(smem_raw + OFF_BR);
  float* sZ   = (float*)(smem_raw + OFF_Z);
  float* sYST = (float*)(smem_raw + OFF_YST);
  float* sK   = (float*)(smem_raw + OFF_K);
  _Float16* sHB  = (_Float16*)(smem_raw + OFF_HB);
  _Float16* sHB2 = (_Float16*)(smem_raw + OFF_HB2);

  const int tid  = threadIdx.x;
  const int lane = tid & 31;
  const int w    = tid >> 5;     // wave id 0..3
  const int ml   = lane & 15;    // col-in-tile (B/C) or row (A)
  const int hh   = lane >> 4;    // half-wave selector
  const int row0 = blockIdx.x * BT;

  // ---------------- init: stage everything into LDS (once) ----------------
  for (int idx = tid; idx < BT * LSEQ * FEAT; idx += NTHR)
    sXT[idx] = (_Float16)x_seq[row0 * LSEQ * FEAT + idx];

  // cde_W2^T pre-swizzled into WMMA-B per-lane layout: B[k][n] = W2[n][k]
  for (int idx = tid; idx < 32 * 2 * 32 * 16; idx += NTHR) {
    int s = idx & 15, ln = (idx >> 4) & 31, kk = (idx >> 9) & 1, tl = idx >> 10;
    int n = 16 * tl + (ln & 15);
    int k = 32 * kk + 16 * (ln >> 4) + s;
    sW2C[idx] = (_Float16)cde_W2[n * HIDN + k];
  }
  for (int idx = tid; idx < 4 * 32 * 16; idx += NTHR) {
    int s = idx & 15, ln = (idx >> 4) & 31, tl = idx >> 9;
    int n = 16 * tl + (ln & 15);
    int k = 16 * (ln >> 4) + s;
    sW1C[idx] = (_Float16)cde_W1[n * INW + k];
    sW1O[idx] = (_Float16)ode_W1[n * INW + k];
  }
  for (int idx = tid; idx < 4 * 2 * 32 * 16; idx += NTHR) {
    int s = idx & 15, ln = (idx >> 4) & 31, kk = (idx >> 9) & 1, tl = idx >> 10;
    int n = 16 * tl + (ln & 15);
    int k = 32 * kk + 16 * (ln >> 4) + s;
    sW2O[idx] = (_Float16)ode_W2[n * HIDN + k];
  }
  for (int idx = tid; idx < 2 * 2 * 32 * 16; idx += NTHR) {
    int s = idx & 15, ln = (idx >> 4) & 31, kk = (idx >> 9) & 1, tl = idx >> 10;
    int n = 16 * tl + (ln & 15);
    int k = 32 * kk + 16 * (ln >> 4) + s;
    sW3O[idx] = (_Float16)ode_W3[n * HIDN + k];
  }
  // effective bias = b1 + one-hot(country) @ W1[:,Z:].T, stored in C-fragment layout
  for (int idx = tid; idx < 4 * 32 * 8; idx += NTHR) {
    int j = idx & 7, ln = (idx >> 3) & 31, wv = idx >> 8;
    int col = 16 * wv + (ln & 15);
    int row = j + 8 * (ln >> 4);
    int ci = country_idx[row0 + row];
    sEBC[idx] = cde_b1[col] + cde_W1[col * INW + ZDIM + ci];
    sEBO[idx] = ode_b1[col] + ode_W1[col * INW + ZDIM + ci];
  }
  for (int idx = tid; idx < 512; idx += NTHR) { sB2C[idx] = cde_b2[idx]; sWR[idx] = Wr[idx]; sZ[idx] = 0.0f; }
  for (int idx = tid; idx < 64;  idx += NTHR) sB2O[idx] = ode_b2[idx];
  for (int idx = tid; idx < 32;  idx += NTHR) sB3O[idx] = ode_b3[idx];
  for (int idx = tid; idx < 16;  idx += NTHR) sBR[idx]  = br[idx];
  __syncthreads();

  // ---------------- persistent register-resident fragments ----------------
  v16h bw1c  = *(const v16h*)(sW1C + (w * 32 + lane) * 16);
  v16h bw1o  = *(const v16h*)(sW1O + (w * 32 + lane) * 16);
  v16h bw2o0 = *(const v16h*)(sW2O + ((w * 2 + 0) * 32 + lane) * 16);
  v16h bw2o1 = *(const v16h*)(sW2O + ((w * 2 + 1) * 32 + lane) * 16);
  const int w3t = w & 1;
  v16h bw3o0 = *(const v16h*)(sW3O + ((w3t * 2 + 0) * 32 + lane) * 16);
  v16h bw3o1 = *(const v16h*)(sW3O + ((w3t * 2 + 1) * 32 + lane) * 16);
  v8f ebc, ebo;
  {
    const float* pc = sEBC + (w * 32 + lane) * 8;
    const float* po = sEBO + (w * 32 + lane) * 8;
#pragma unroll
    for (int j = 0; j < 8; ++j) { ebc[j] = pc[j]; ebo[j] = po[j]; }
  }

  // A-fragment (16x32, K=32) from f32 stage buffer [16][32]
  auto afragY = [&]() -> v16h {
    const float* yr = sYST + ml * ZDIM;
    v16h a;
#pragma unroll
    for (int s = 0; s < 8; ++s) a[s]     = (_Float16)yr[16 * hh + s];
#pragma unroll
    for (int s = 0; s < 8; ++s) a[8 + s] = (_Float16)yr[16 * hh + 8 + s];
    return a;
  };
  // A-fragment (16x32, K-slab kk of K=64) from f16 buffer [16][64]
  auto afragH = [&](const _Float16* H, int kk) -> v16h {
    const _Float16* p = H + ml * HIDN + 32 * kk;
    v16h a;
#pragma unroll
    for (int s = 0; s < 8; ++s) a[s]     = p[16 * hh + s];
#pragma unroll
    for (int s = 0; s < 8; ++s) a[8 + s] = p[16 * hh + 8 + s];
    return a;
  };

  // ---------------- CDE vector field: k[kslot] = f_cde(tc, sYST) ----------------
  auto cde_vf = [&](float tc, int kslot) {
    // h = tanh(z@W1z.T + ebias)   (wave w computes cols 16w..16w+15)
    v16h a = afragY();
    v8f acc = ebc;
    acc = wmma16(a, bw1c, acc);
    const int col = 16 * w + ml;
#pragma unroll
    for (int j = 0; j < 8; ++j) sHB[(j + 8 * hh) * HIDN + col] = (_Float16)ftanh(acc[j]);
    __syncthreads();
    v16h a0 = afragH(sHB, 0);
    v16h a1 = afragH(sHB, 1);

    // cubic-hermite path derivative dX[row][f] in C-fragment layout
    int i = (int)tc;
    if (i > LSEQ - 2) i = LSEQ - 2;
    float s = tc - (float)i;
    float h00 = 6.0f * s * s - 6.0f * s;
    float h10 = 3.0f * s * s - 4.0f * s + 1.0f;
    float h11 = 3.0f * s * s - 2.0f * s;
    int im1 = (i >= 1) ? (i - 1) : 0;
    float dxv[8];
#pragma unroll
    for (int j = 0; j < 8; ++j) {
      const _Float16* xr = sXT + (j + 8 * hh) * (LSEQ * FEAT) + ml;
      float y0 = (float)xr[i * FEAT];
      float y1 = (float)xr[(i + 1) * FEAT];
      float ym = (float)xr[im1 * FEAT];
      float d1 = y1 - y0;
      float d0 = (i >= 1) ? (y0 - ym) : d1;
      dxv[j] = h00 * (y0 - y1) + h10 * d0 + h11 * d1;
    }

    // mat = h @ W2.T + b2, tile n == z; contract over f with shuffle butterfly
    float* kdst = sK + kslot * 512;
#pragma unroll
    for (int tl = 0; tl < 8; ++tl) {
      int tg = 8 * w + tl;                 // z value this tile produces
      float bb = sB2C[16 * tg + ml];
      v8f acc2;
#pragma unroll
      for (int j = 0; j < 8; ++j) acc2[j] = bb;
      v16h b0 = *(const v16h*)(sW2C + ((tg * 2 + 0) * 32 + lane) * 16);
      acc2 = wmma16(a0, b0, acc2);
      v16h b1 = *(const v16h*)(sW2C + ((tg * 2 + 1) * 32 + lane) * 16);
      acc2 = wmma16(a1, b1, acc2);
      float p[8];
#pragma unroll
      for (int j = 0; j < 8; ++j) p[j] = acc2[j] * dxv[j];
#pragma unroll
      for (int m = 1; m < 16; m <<= 1) {
#pragma unroll
        for (int j = 0; j < 8; ++j) p[j] += __shfl_xor(p[j], m, 32);
      }
      if (ml == 0) {
#pragma unroll
        for (int j = 0; j < 8; ++j) kdst[(j + 8 * hh) * ZDIM + tg] = p[j];
      }
    }
    __syncthreads();
  };

  // ---------------- ODE vector field: k[kslot] = f_ode(sYST) ----------------
  auto ode_vf = [&](int kslot) {
    v16h a = afragY();
    v8f acc = ebo;
    acc = wmma16(a, bw1o, acc);
    const int col = 16 * w + ml;
#pragma unroll
    for (int j = 0; j < 8; ++j) sHB[(j + 8 * hh) * HIDN + col] = (_Float16)ftanh(acc[j]);
    __syncthreads();
    {
      v16h a0 = afragH(sHB, 0);
      v16h a1 = afragH(sHB, 1);
      float bb = sB2O[col];
      v8f acc2;
#pragma unroll
      for (int j = 0; j < 8; ++j) acc2[j] = bb;
      acc2 = wmma16(a0, bw2o0, acc2);
      acc2 = wmma16(a1, bw2o1, acc2);
#pragma unroll
      for (int j = 0; j < 8; ++j) sHB2[(j + 8 * hh) * HIDN + col] = (_Float16)ftanh(acc2[j]);
    }
    __syncthreads();
    if (w < 2) {
      v16h c0 = afragH(sHB2, 0);
      v16h c1 = afragH(sHB2, 1);
      float b3 = sB3O[16 * w + ml];
      v8f acc3;
#pragma unroll
      for (int j = 0; j < 8; ++j) acc3[j] = b3;
      acc3 = wmma16(c0, bw3o0, acc3);
      acc3 = wmma16(c1, bw3o1, acc3);
      float* kdst = sK + kslot * 512;
#pragma unroll
      for (int j = 0; j < 8; ++j) kdst[(j + 8 * hh) * ZDIM + 16 * w + ml] = acc3[j];
    }
    __syncthreads();
  };

  // ---------------- RK stage prep / combine (float4 per thread) ----------------
  auto prep = [&](int cnt, float c1, float c2, float c3, float c4, float c5) {
    int i0 = tid * 4;
    float4 y = *(const float4*)(sZ + i0);
    if (cnt >= 1) { float4 k = *(const float4*)(sK + 0 * 512 + i0); y.x += c1 * k.x; y.y += c1 * k.y; y.z += c1 * k.z; y.w += c1 * k.w; }
    if (cnt >= 2) { float4 k = *(const float4*)(sK + 1 * 512 + i0); y.x += c2 * k.x; y.y += c2 * k.y; y.z += c2 * k.z; y.w += c2 * k.w; }
    if (cnt >= 3) { float4 k = *(const float4*)(sK + 2 * 512 + i0); y.x += c3 * k.x; y.y += c3 * k.y; y.z += c3 * k.z; y.w += c3 * k.w; }
    if (cnt >= 4) { float4 k = *(const float4*)(sK + 3 * 512 + i0); y.x += c4 * k.x; y.y += c4 * k.y; y.z += c4 * k.z; y.w += c4 * k.w; }
    if (cnt >= 5) { float4 k = *(const float4*)(sK + 4 * 512 + i0); y.x += c5 * k.x; y.y += c5 * k.y; y.z += c5 * k.z; y.w += c5 * k.w; }
    *(float4*)(sYST + i0) = y;
    __syncthreads();
  };
  auto combine = [&]() {
    int i0 = tid * 4;
    float4 y = *(const float4*)(sZ + i0);
    const float bc[6] = {Bc1, Bc2, Bc3, Bc4, Bc5, Bc6};
#pragma unroll
    for (int j = 0; j < 6; ++j) {
      float4 k = *(const float4*)(sK + j * 512 + i0);
      y.x += bc[j] * k.x; y.y += bc[j] * k.y; y.z += bc[j] * k.z; y.w += bc[j] * k.w;
    }
    *(float4*)(sZ + i0) = y;
    __syncthreads();
  };

  // ---------------- encoder: 63 Tsit5 steps of the CDE field ----------------
  for (int step = 0; step < LSEQ - 1; ++step) {
    float t = (float)step;
    prep(0, 0, 0, 0, 0, 0);                  cde_vf(t, 0);
    prep(1, A21, 0, 0, 0, 0);                cde_vf(t + C2c, 1);
    prep(2, A31, A32, 0, 0, 0);              cde_vf(t + C3c, 2);
    prep(3, A41, A42, A43, 0, 0);            cde_vf(t + C4c, 3);
    prep(4, A51, A52, A53, A54, 0);          cde_vf(t + C5c, 4);
    prep(5, A61, A62, A63, A64, A65);        cde_vf(t + C6c, 5);
    combine();
  }

  // ---------------- decoder: 12 Tsit5 steps of the ODE field + readout ----------------
  for (int hstep = 0; hstep < HORZ; ++hstep) {
    prep(0, 0, 0, 0, 0, 0);                  ode_vf(0);
    prep(1, A21, 0, 0, 0, 0);                ode_vf(1);
    prep(2, A31, A32, 0, 0, 0);              ode_vf(2);
    prep(3, A41, A42, A43, 0, 0);            ode_vf(3);
    prep(4, A51, A52, A53, A54, 0);          ode_vf(4);
    prep(5, A61, A62, A63, A64, A65);        ode_vf(5);
    combine();
    if (tid < BT * 3) {
      int r = tid / 3, kk = tid % 3;
      int f = (kk == 0) ? 0 : ((kk == 1) ? 3 : 7);
      float accv = sBR[f];
#pragma unroll
      for (int z = 0; z < ZDIM; ++z) accv += sZ[r * ZDIM + z] * sWR[f * ZDIM + z];
      out[(row0 + r) * (HORZ * 3) + hstep * 3 + kk] = accv;
    }
    __syncthreads();
  }
}

extern "C" void kernel_launch(void* const* d_in, const int* in_sizes, int n_in,
                              void* d_out, int out_size, void* d_ws, size_t ws_size,
                              hipStream_t stream) {
  (void)in_sizes; (void)n_in; (void)out_size; (void)d_ws; (void)ws_size;
  const float* x_seq  = (const float*)d_in[0];
  const int*   cidx   = (const int*)d_in[1];
  const float* cde_W1 = (const float*)d_in[2];
  const float* cde_b1 = (const float*)d_in[3];
  const float* cde_W2 = (const float*)d_in[4];
  const float* cde_b2 = (const float*)d_in[5];
  const float* ode_W1 = (const float*)d_in[6];
  const float* ode_b1 = (const float*)d_in[7];
  const float* ode_W2 = (const float*)d_in[8];
  const float* ode_b2 = (const float*)d_in[9];
  const float* ode_W3 = (const float*)d_in[10];
  const float* ode_b3 = (const float*)d_in[11];
  const float* Wr     = (const float*)d_in[12];
  const float* br     = (const float*)d_in[13];
  float* out = (float*)d_out;

  dim3 grid(BATCH / BT);   // 32 persistent workgroups, one 16-row batch tile each
  dim3 block(NTHR);        // 4 waves (wave32) per workgroup
  ncde_persistent_kernel<<<grid, block, SMEM_BYTES, stream>>>(
      x_seq, cidx, cde_W1, cde_b1, cde_W2, cde_b2,
      ode_W1, ode_b1, ode_W2, ode_b2, ode_W3, ode_b3, Wr, br, out);
}
